// MambaForecaster_50637664420483
// MI455X (gfx1250) — compile-verified, matrix operand order checked by reading, not run on hardware
//
#include <hip/hip_runtime.h>
#include <math.h>
#include <stdint.h>

// ---------------- model constants (match reference) ----------------
#define D_IN     8
#define D_MODEL  256
#define N_LAYERS 2
#define HORIZON  24
#define D_STATE  64
#define D_CONV   4
#define D_INNER  512        // EXPAND * D_MODEL
#define DT_RANK  16         // (D_MODEL + 15) // 16
#define BB       4
#define TT       1024
#define NROW     (BB * TT)  // 4096
#define XDBL_N   (DT_RANK + 2 * D_STATE)  // 144
#define SCAN_CH  16         // timesteps per async-staged chunk

typedef __attribute__((ext_vector_type(16))) _Float16 v16h;
typedef __attribute__((ext_vector_type(8)))  _Float16 v8h;
typedef __attribute__((ext_vector_type(8)))  float    v8f;

// ---------------- f32 -> f16 conversion ----------------
__global__ void k_cvt_f16(const float* __restrict__ src, _Float16* __restrict__ dst, int n) {
    int i = blockIdx.x * blockDim.x + threadIdx.x;
    if (i < n) dst[i] = (_Float16)src[i];
}

// ---------------- input projection: h = x @ in_proj_w.T + b ----------------
__global__ void k_inproj(const float* __restrict__ x, const float* __restrict__ w,
                         const float* __restrict__ bias,
                         float* __restrict__ h, _Float16* __restrict__ h16) {
    int i = blockIdx.x * blockDim.x + threadIdx.x;     // over NROW*D_MODEL
    int row = i >> 8;
    int d   = i & (D_MODEL - 1);
    const float* xr = x + (size_t)row * D_IN;
    const float* wr = w + (size_t)d * D_IN;
    float acc = bias[d];
#pragma unroll
    for (int k = 0; k < D_IN; ++k) acc += xr[k] * wr[k];
    h[i] = acc;
    h16[i] = (_Float16)acc;
}

// ---------------- WMMA GEMM: C[M,N] = A[M,K] @ W[N,K]^T ----------------
// One wave computes a 16 x (16*NT) output tile with v_wmma_f32_16x16x32_f16.
// Per-lane fragment layout (CDNA5 ISA, 16-bit A 16x32):
//   lane<16 : row=lane,    K = kc+0..7  and kc+16..23
//   lane>=16: row=lane-16, K = kc+8..15 and kc+24..31
template <int NT>
__global__ void k_wmma_gemm(const _Float16* __restrict__ A,
                            const _Float16* __restrict__ W,
                            float* __restrict__ C,
                            _Float16* __restrict__ C16,   // optional (may be null)
                            int M, int N, int K) {
    const int wave = threadIdx.x >> 5;
    const int lane = threadIdx.x & 31;
    const int half = lane >> 4;
    const int lr   = lane & 15;
    const int m0 = (blockIdx.y * (blockDim.x >> 5) + wave) * 16;
    const int n0 = blockIdx.x * (16 * NT);
    if (m0 >= M) return;

    v8f acc[NT] = {};
    const _Float16* arow = A + (size_t)(m0 + lr) * K;

    for (int kc = 0; kc < K; kc += 32) {
        const int kb = kc + half * 8;
        union { v16h v; v8h h[2]; } a;
        a.h[0] = *(const v8h*)(arow + kb);
        a.h[1] = *(const v8h*)(arow + kb + 16);
#pragma unroll
        for (int nt = 0; nt < NT; ++nt) {
            const _Float16* wrow = W + (size_t)(n0 + nt * 16 + lr) * K;
            union { v16h v; v8h h[2]; } b;
            b.h[0] = *(const v8h*)(wrow + kb);
            b.h[1] = *(const v8h*)(wrow + kb + 16);
            acc[nt] = __builtin_amdgcn_wmma_f32_16x16x32_f16(
                false, a.v, false, b.v, (short)0, acc[nt], false, false);
        }
    }

    // C/D layout: VGPR j -> (row m0+j+half*8, col n0+nt*16+lr)
#pragma unroll
    for (int nt = 0; nt < NT; ++nt) {
        const int col = n0 + nt * 16 + lr;
#pragma unroll
        for (int j = 0; j < 8; ++j) {
            const int row = m0 + j + half * 8;
            const float v = acc[nt][j];
            C[(size_t)row * N + col] = v;
            if (C16) C16[(size_t)row * N + col] = (_Float16)v;
        }
    }
}

// ---------------- causal depthwise conv (width 4) + SiLU ----------------
__global__ void k_conv_silu(const float* __restrict__ xz,   // [NROW, 2*D_INNER]
                            const float* __restrict__ cw,   // [D_INNER, D_CONV]
                            const float* __restrict__ cb,   // [D_INNER]
                            float* __restrict__ xs, _Float16* __restrict__ xs16) {
    int i = blockIdx.x * blockDim.x + threadIdx.x;  // NROW*D_INNER
    int row = i >> 9;
    int c   = i & (D_INNER - 1);
    int t   = row & (TT - 1);
    const float* w = cw + (size_t)c * D_CONV;
    float acc = cb[c];
#pragma unroll
    for (int k = 0; k < D_CONV; ++k) {
        int tt = t - (D_CONV - 1) + k;
        if (tt >= 0)
            acc += w[k] * xz[(size_t)(row - (D_CONV - 1) + k) * (2 * D_INNER) + c];
    }
    float s = acc / (1.0f + __expf(-acc));  // SiLU
    xs[i] = s;
    xs16[i] = (_Float16)s;
}

// ---------------- dt = softplus(dt_r @ dtproj_w.T + b), K = 16 ----------------
__global__ void k_dt(const float* __restrict__ xdbl,  // [NROW, 144], dt_r = [:, :16]
                     const float* __restrict__ dtw,   // [D_INNER, DT_RANK]
                     const float* __restrict__ dtb,   // [D_INNER]
                     float* __restrict__ dt) {
    int i = blockIdx.x * blockDim.x + threadIdx.x;  // NROW*D_INNER
    int row = i >> 9;
    int c   = i & (D_INNER - 1);
    const float* xr = xdbl + (size_t)row * XDBL_N;
    const float* wr = dtw + (size_t)c * DT_RANK;
    float acc = dtb[c];
#pragma unroll
    for (int r = 0; r < DT_RANK; ++r) acc += xr[r] * wr[r];
    dt[i] = (acc > 20.0f) ? acc : log1pf(__expf(acc));
}

// ---------------- selective scan (async-LDS double-buffered) ----------------
// Block = 8 waves, one wave per channel c (all waves share batch b).
// 64 states per (b,c) = 2 per lane, resident in registers for all T.
// A 16-timestep chunk of (B,C) rows (shared by all 8 waves) plus this block's
// dt/xs slices is staged into LDS with GLOBAL_LOAD_ASYNC_TO_LDS_{B128,B32}
// (ASYNCcnt), double-buffered so the copy of chunk k+1 overlaps compute of k.
// Every thread issues exactly 3 async ops per chunk, so s_wait_asynccnt 3
// after issuing chunk k+1 guarantees chunk k has landed (in-order completion).
__global__ void k_scan(const float* __restrict__ dt,    // [NROW, D_INNER]
                       const float* __restrict__ xs,    // [NROW, D_INNER]
                       const float* __restrict__ xdbl,  // [NROW, 144] (B at 16, C at 80)
                       const float* __restrict__ Alog,  // [D_INNER, D_STATE] (this layer)
                       float* __restrict__ yscan) {
    __shared__ float sh_xd[2][SCAN_CH * 128];  // per row: B[64] | C[64]
    __shared__ float sh_dt[2][SCAN_CH * 8];
    __shared__ float sh_xs[2][SCAN_CH * 8];

    const int tid  = threadIdx.x;
    const int wave = tid >> 5;
    const int lane = tid & 31;
    const int b    = blockIdx.x >> 6;                 // 64 blocks per batch
    const int c0   = (blockIdx.x * 8) & (D_INNER - 1);
    const int c    = c0 + wave;
    const int n0   = lane * 2;
    const size_t rowbase = (size_t)b * TT;

    float A0 = -__expf(Alog[(size_t)c * D_STATE + n0]);
    float A1 = -__expf(Alog[(size_t)c * D_STATE + n0 + 1]);
    float h0 = 0.0f, h1 = 0.0f;

    auto issue_chunk = [&](int t0, int bufi) {
        // (B,C) region of x_dbl rows t0..t0+15: 512 x b128 per chunk, 2/thread
#pragma unroll
        for (int j = 0; j < 2; ++j) {
            int idx = tid + j * 256;          // 0..511
            int r   = idx >> 5;               // row in chunk
            int seg = idx & 31;               // 16B segment in row
            unsigned long long g =
                (unsigned long long)(const void*)(xdbl + (rowbase + t0 + r) * XDBL_N + DT_RANK)
                + (unsigned)seg * 16u;
            unsigned l = (unsigned)(uintptr_t)&sh_xd[bufi][r * 128] + (unsigned)seg * 16u;
            asm volatile("global_load_async_to_lds_b128 %0, %1, off"
                         :: "v"(l), "v"(g) : "memory");
        }
        // dt (threads 0..127) / xs (threads 128..255) slices: 1 b32/thread
        {
            int r = (tid & 127) >> 3;
            int w = tid & 7;
            const float* src = (tid < 128)
                ? (dt + (rowbase + t0 + r) * D_INNER + c0 + w)
                : (xs + (rowbase + t0 + r) * D_INNER + c0 + w);
            float* dl = (tid < 128) ? &sh_dt[bufi][r * 8 + w] : &sh_xs[bufi][r * 8 + w];
            unsigned long long g = (unsigned long long)(const void*)src;
            unsigned l = (unsigned)(uintptr_t)dl;
            asm volatile("global_load_async_to_lds_b32 %0, %1, off"
                         :: "v"(l), "v"(g) : "memory");
        }
    };

    const int NCH = TT / SCAN_CH;
    issue_chunk(0, 0);

    for (int ch = 0; ch < NCH; ++ch) {
        const int bufi = ch & 1;
        if (ch + 1 < NCH) {
            issue_chunk((ch + 1) * SCAN_CH, (ch + 1) & 1);
            asm volatile("s_wait_asynccnt 0x3" ::: "memory");  // chunk ch landed
        } else {
            asm volatile("s_wait_asynccnt 0x0" ::: "memory");
        }
        __syncthreads();  // make all threads' LDS writes visible

#pragma unroll 4
        for (int r = 0; r < SCAN_CH; ++r) {
            float dtv = sh_dt[bufi][r * 8 + wave];
            float xv  = sh_xs[bufi][r * 8 + wave];
            float2 Bv = *(const float2*)&sh_xd[bufi][r * 128 + n0];
            float2 Cv = *(const float2*)&sh_xd[bufi][r * 128 + 64 + n0];
            float dbu = dtv * xv;
            h0 = __expf(dtv * A0) * h0 + dbu * Bv.x;
            h1 = __expf(dtv * A1) * h1 + dbu * Bv.y;
            float p = h0 * Cv.x + h1 * Cv.y;
#pragma unroll
            for (int off = 16; off >= 1; off >>= 1) p += __shfl_xor(p, off, 32);
            if (lane == 0)
                yscan[(rowbase + ch * SCAN_CH + r) * D_INNER + c] = p;
        }
        __syncthreads();  // chunk consumed before its buffer is re-issued
    }
}

// ---------------- gate: y16 = (yscan + xs*D) * silu(z) ----------------
__global__ void k_gate(const float* __restrict__ yscan, const float* __restrict__ xs,
                       const float* __restrict__ xz, const float* __restrict__ Dp,
                       _Float16* __restrict__ y16) {
    int i = blockIdx.x * blockDim.x + threadIdx.x;  // NROW*D_INNER
    int row = i >> 9;
    int c   = i & (D_INNER - 1);
    float z = xz[(size_t)row * (2 * D_INNER) + D_INNER + c];
    float g = z / (1.0f + __expf(-z));
    float y = (yscan[i] + xs[i] * Dp[c]) * g;
    y16[i] = (_Float16)y;
}

// ---------------- head: LN(last t) -> GELU MLP -> [B, 24] ----------------
__global__ void k_head(const float* __restrict__ h, const float* __restrict__ g,
                       const float* __restrict__ be,
                       const float* __restrict__ w1, const float* __restrict__ b1,
                       const float* __restrict__ w2, const float* __restrict__ b2,
                       float* __restrict__ out) {
    __shared__ float red[D_MODEL];
    __shared__ float ln[D_MODEL];
    __shared__ float hid[D_MODEL];
    int b = blockIdx.x, tid = threadIdx.x;

    float v = h[((size_t)b * TT + TT - 1) * D_MODEL + tid];
    red[tid] = v; __syncthreads();
    for (int s = D_MODEL / 2; s > 0; s >>= 1) {
        if (tid < s) red[tid] += red[tid + s];
        __syncthreads();
    }
    float mu = red[0] / (float)D_MODEL;
    __syncthreads();
    float d = v - mu;
    red[tid] = d * d; __syncthreads();
    for (int s = D_MODEL / 2; s > 0; s >>= 1) {
        if (tid < s) red[tid] += red[tid + s];
        __syncthreads();
    }
    float var = red[0] / (float)D_MODEL;
    __syncthreads();
    ln[tid] = d * rsqrtf(var + 1e-5f) * g[tid] + be[tid];
    __syncthreads();

    float acc = b1[tid];
    const float* w1r = w1 + (size_t)tid * D_MODEL;
    for (int k = 0; k < D_MODEL; ++k) acc += ln[k] * w1r[k];
    hid[tid] = 0.5f * acc * (1.0f + erff(acc * 0.70710678118f));  // exact GELU
    __syncthreads();

    if (tid < HORIZON) {
        float o = b2[tid];
        const float* w2r = w2 + (size_t)tid * D_MODEL;
        for (int k = 0; k < D_MODEL; ++k) o += hid[k] * w2r[k];
        out[b * HORIZON + tid] = o;
    }
}

// ---------------- launch ----------------
extern "C" void kernel_launch(void* const* d_in, const int* in_sizes, int n_in,
                              void* d_out, int out_size, void* d_ws, size_t ws_size,
                              hipStream_t stream) {
    (void)in_sizes; (void)n_in; (void)out_size; (void)ws_size;
    const float* x     = (const float*)d_in[0];
    const float* inw   = (const float*)d_in[1];
    const float* inb   = (const float*)d_in[2];
    const float* binw  = (const float*)d_in[3];   // [2,1024,256]
    const float* bcw   = (const float*)d_in[4];   // [2,512,4]
    const float* bcb   = (const float*)d_in[5];   // [2,512]
    const float* bxw   = (const float*)d_in[6];   // [2,144,512]
    const float* bdtw  = (const float*)d_in[7];   // [2,512,16]
    const float* bdtb  = (const float*)d_in[8];   // [2,512]
    const float* bAlog = (const float*)d_in[9];   // [2,512,64]
    const float* bD    = (const float*)d_in[10];  // [2,512]
    const float* boutw = (const float*)d_in[11];  // [2,256,512]
    const float* ng    = (const float*)d_in[12];
    const float* nbias = (const float*)d_in[13];
    const float* hw1   = (const float*)d_in[14];
    const float* hb1   = (const float*)d_in[15];
    const float* hw2   = (const float*)d_in[16];
    const float* hb2   = (const float*)d_in[17];
    float* out = (float*)d_out;

    // workspace carve-up (256B aligned)
    char* ws = (char*)d_ws;
    size_t off = 0;
    auto carve = [&](size_t bytes) { size_t o = off; off += (bytes + 255) & ~(size_t)255; return o; };
    float*    h      = (float*)   (ws + carve((size_t)NROW * D_MODEL * 4));
    _Float16* h16    = (_Float16*)(ws + carve((size_t)NROW * D_MODEL * 2));
    float*    xz     = (float*)   (ws + carve((size_t)NROW * 2 * D_INNER * 4));
    float*    xs     = (float*)   (ws + carve((size_t)NROW * D_INNER * 4));
    _Float16* xs16   = (_Float16*)(ws + carve((size_t)NROW * D_INNER * 2));
    float*    xdbl   = (float*)   (ws + carve((size_t)NROW * XDBL_N * 4));
    float*    dtb_   = (float*)   (ws + carve((size_t)NROW * D_INNER * 4));
    float*    yscan  = (float*)   (ws + carve((size_t)NROW * D_INNER * 4));
    _Float16* y16    = (_Float16*)(ws + carve((size_t)NROW * D_INNER * 2));
    _Float16* w_in16 = (_Float16*)(ws + carve((size_t)N_LAYERS * 2 * D_INNER * D_MODEL * 2));
    _Float16* w_x16  = (_Float16*)(ws + carve((size_t)N_LAYERS * XDBL_N * D_INNER * 2));
    _Float16* w_o16  = (_Float16*)(ws + carve((size_t)N_LAYERS * D_MODEL * D_INNER * 2));

    // convert all layer weights once
    {
        int n1 = N_LAYERS * 2 * D_INNER * D_MODEL;   // 524288
        int n2 = N_LAYERS * XDBL_N * D_INNER;        // 147456
        int n3 = N_LAYERS * D_MODEL * D_INNER;       // 262144
        k_cvt_f16<<<(n1 + 255) / 256, 256, 0, stream>>>(binw, w_in16, n1);
        k_cvt_f16<<<(n2 + 255) / 256, 256, 0, stream>>>(bxw,  w_x16,  n2);
        k_cvt_f16<<<(n3 + 255) / 256, 256, 0, stream>>>(boutw, w_o16, n3);
    }

    // input projection
    k_inproj<<<(NROW * D_MODEL) / 256, 256, 0, stream>>>(x, inw, inb, h, h16);

    const int EW = NROW * D_INNER;  // elementwise over [NROW, D_INNER]

    for (int l = 0; l < N_LAYERS; ++l) {
        const _Float16* wi = w_in16 + (size_t)l * 2 * D_INNER * D_MODEL;
        const _Float16* wx = w_x16  + (size_t)l * XDBL_N * D_INNER;
        const _Float16* wo = w_o16  + (size_t)l * D_MODEL * D_INNER;

        // xz = h @ inproj^T : M=4096 N=1024 K=256, 16x64 tiles/wave
        k_wmma_gemm<4><<<dim3(2 * D_INNER / 64, NROW / 16 / 8), 256, 0, stream>>>(
            h16, wi, xz, nullptr, NROW, 2 * D_INNER, D_MODEL);

        // causal conv + SiLU
        k_conv_silu<<<EW / 256, 256, 0, stream>>>(
            xz, bcw + (size_t)l * D_INNER * D_CONV, bcb + (size_t)l * D_INNER, xs, xs16);

        // x_dbl = xs @ xproj^T : M=4096 N=144 K=512, 16x48 tiles/wave
        k_wmma_gemm<3><<<dim3(XDBL_N / 48, NROW / 16 / 8), 256, 0, stream>>>(
            xs16, wx, xdbl, nullptr, NROW, XDBL_N, D_INNER);

        // dt = softplus(dt_r @ dtproj^T + b)
        k_dt<<<EW / 256, 256, 0, stream>>>(
            xdbl, bdtw + (size_t)l * D_INNER * DT_RANK, bdtb + (size_t)l * D_INNER, dtb_);

        // selective scan: 2048 waves, states in registers, async-LDS staging
        k_scan<<<(BB * D_INNER) / 8, 256, 0, stream>>>(
            dtb_, xs, xdbl, bAlog + (size_t)l * D_INNER * D_STATE, yscan);

        // gate
        k_gate<<<EW / 256, 256, 0, stream>>>(yscan, xs, xz, bD + (size_t)l * D_INNER, y16);

        // h = y @ outproj^T : M=4096 N=256 K=512 (f32 + f16 for next layer)
        k_wmma_gemm<4><<<dim3(D_MODEL / 64, NROW / 16 / 8), 256, 0, stream>>>(
            y16, wo, h, h16, NROW, D_MODEL, D_INNER);
    }

    // LN on last timestep + GELU MLP head
    k_head<<<BB, D_MODEL, 0, stream>>>(h, ng, nbias, hw1, hb1, hw2, hb2, out);
}
